// GraphAttentionLayer_75866302317208
// MI455X (gfx1250) — matrix operand order repaired and checked by reading, not compiled, not run on hardware
//
#include <hip/hip_runtime.h>
#include <hip/hip_bf16.h>

// GAT layer on MI455X (gfx1250, wave32, WMMA).
// B=8, N=2048, F_IN=F_OUT=256.
#define BATCH 8
#define NNODE 2048
#define FDIM  256

typedef __attribute__((ext_vector_type(8)))  __bf16 v8bf;
typedef __attribute__((ext_vector_type(16))) __bf16 v16bf;
typedef __attribute__((ext_vector_type(8)))  float  v8f;

static __device__ __forceinline__ v16bf concat8(v8bf a, v8bf b) {
  return __builtin_shufflevector(a, b, 0,1,2,3,4,5,6,7,8,9,10,11,12,13,14,15);
}
// 16 bf16 from two 8-element runs (A-fragment interleaved K pattern / LDS tiles)
static __device__ __forceinline__ v16bf ld_bf16_8x2(const __bf16* p0, const __bf16* p1) {
  return concat8(*(const v8bf*)p0, *(const v8bf*)p1);
}
// convert two 8-float runs -> v16bf (A fragment from f32 memory)
static __device__ __forceinline__ v16bf cvt_f32_8x2(const float* p0, const float* p1) {
  v16bf r;
#pragma unroll
  for (int t = 0; t < 8; ++t) { r[t] = (__bf16)p0[t]; r[t + 8] = (__bf16)p1[t]; }
  return r;
}

static __device__ __forceinline__ v8f wmma_bf16(v16bf a, v16bf b, v8f c) {
  return __builtin_amdgcn_wmma_f32_16x16x32_bf16(false, a, false, b, (short)0, c,
                                                 false, false);
}

// ---------------------------------------------------------------------------
// Kernel 0: one-time W f32 -> bf16 conversion (128 KB, L2-resident afterwards).
// Removes per-k-step v_cvt_pk_bf16_f32 + WMMA->VALU WAR nops in kernel 1.
// ---------------------------------------------------------------------------
__global__ void __launch_bounds__(256)
wcvt_kernel(const float* __restrict__ Ww, __bf16* __restrict__ Wbf)
{
  const int i = (blockIdx.x * 256 + threadIdx.x) * 4;
#pragma unroll
  for (int t = 0; t < 4; ++t) Wbf[i + t] = (__bf16)Ww[i + t];
}

// ---------------------------------------------------------------------------
// Kernel 1: Wh = h @ W^T + bias, written TRANSPOSED as bf16: wh_t[b][o][n].
// Block: 256 thr (8 waves) -> 64 rows x 256 features. Wave = 16 rows x 128 feats.
// K = 256, stepped by 32 (v_wmma_f32_16x16x32_bf16).
// ---------------------------------------------------------------------------
__global__ void __launch_bounds__(256)
wh_gemm_kernel(const float* __restrict__ h,     // [B, N, F]
               const __bf16* __restrict__ Wbf,  // [F_out, F_in] row-major bf16
               const float* __restrict__ Wb,    // [F_out]
               __bf16* __restrict__ wh_t)       // [B, F_out, N]
{
  const int b    = blockIdx.y;
  const int i0   = blockIdx.x * 64;
  const int tid  = threadIdx.x;
  const int lane = tid & 31;
  const int wave = tid >> 5;
  const int rg   = wave & 3;   // row group (16 rows)
  const int fg   = wave >> 2;  // feature group (128 features)

  const float* hB = h + (size_t)b * NNODE * FDIM;

  __shared__ __align__(16) __bf16 ldsT[FDIM][72];  // transposed staging [o][n], padded

  v8f acc[8] = {};  // 8 C-tiles of 16x16 f32

  const int arow = i0 + rg * 16 + (lane & 15);     // global row for A frag
  const int akb  = (lane < 16) ? 0 : 8;            // A interleaved K base
  const int bkb  = (lane < 16) ? 0 : 8;            // B interleaved K base

  for (int k0 = 0; k0 < FDIM; k0 += 32) {
    const float* pa = hB + (size_t)arow * FDIM + k0 + akb;
    v16bf afrag = cvt_f32_8x2(pa, pa + 16);  // h converted on the fly (read once)
#pragma unroll
    for (int ct = 0; ct < 8; ++ct) {
      const int o = fg * 128 + ct * 16 + (lane & 15);
      const __bf16* pb = Wbf + (size_t)o * FDIM + k0 + bkb;  // B[k][o] = W[o][k]
      v16bf bfrag = ld_bf16_8x2(pb, pb + 16);
      acc[ct] = wmma_bf16(afrag, bfrag, acc[ct]);
    }
  }

  // bias + transpose into LDS (C layout: VGPR v -> row v + 8*(lane>>4), col lane&15)
#pragma unroll
  for (int ct = 0; ct < 8; ++ct) {
    const int ocol   = fg * 128 + ct * 16 + (lane & 15);
    const float bias = Wb[ocol];
    const int rbase  = rg * 16 + 8 * (lane >> 4);
#pragma unroll
    for (int v = 0; v < 8; ++v)
      ldsT[ocol][rbase + v] = (__bf16)(acc[ct][v] + bias);
  }
  __syncthreads();

  // coalesced write-out: thread tid owns output feature o = tid, 64 nodes
  __bf16* dst = wh_t + ((size_t)b * FDIM + tid) * NNODE + i0;
  const __bf16* src = &ldsT[tid][0];
#pragma unroll
  for (int n = 0; n < 64; n += 8)
    *(uint4*)(dst + n) = *(const uint4*)(src + n);
}

// ---------------------------------------------------------------------------
// Kernel 2: f1[b,n] = a1 . Wh[b,n,:] + a1b ; f2 likewise (single pass over wh_t).
// ---------------------------------------------------------------------------
__global__ void __launch_bounds__(256)
f12_kernel(const __bf16* __restrict__ wh_t,
           const float* __restrict__ a1w, const float* __restrict__ a1b,
           const float* __restrict__ a2w, const float* __restrict__ a2b,
           float* __restrict__ f1, float* __restrict__ f2)
{
  const int b = blockIdx.y;
  const int n = blockIdx.x * 256 + threadIdx.x;
  const __bf16* base = wh_t + (size_t)b * FDIM * NNODE + n;
  float s1 = 0.f, s2 = 0.f;
#pragma unroll 4
  for (int o = 0; o < FDIM; ++o) {
    const float w = (float)base[(size_t)o * NNODE];  // coalesced across threads
    s1 += a1w[o] * w;
    s2 += a2w[o] * w;
  }
  f1[b * NNODE + n] = s1 + a1b[0];
  f2[b * NNODE + n] = s2 + a2b[0];
}

// ---------------------------------------------------------------------------
// Kernel 3: column softmax denominators.
// s[b,j] = sum_i (adj[b,i,j] > 0) * exp(relu(f1[i] + f2[j])).
// relu >= 0 and exp(-9e15) underflows to 0, so no max-subtraction is needed;
// s==0 <=> fully-masked column (uniform fallback applied in kernel 4).
// Block: 256 thr -> 64 columns, 4-way row split, LDS reduce.
// ---------------------------------------------------------------------------
__global__ void __launch_bounds__(256)
stats_kernel(const int* __restrict__ adj, const float* __restrict__ f1,
             const float* __restrict__ f2, float* __restrict__ s_out)
{
  const int b   = blockIdx.y;
  const int j0  = blockIdx.x * 64;
  const int tid = threadIdx.x;
  const int col = tid & 63;
  const int rq  = tid >> 6;  // 0..3
  const int j   = j0 + col;

  const float fj = f2[b * NNODE + j];
  const int* adjB = adj + (size_t)b * NNODE * NNODE;
  const float* f1B = f1 + b * NNODE;

  float acc = 0.f;
  for (int i = rq; i < NNODE; i += 4) {
    const int a  = adjB[(size_t)i * NNODE + j];
    const float e = fmaxf(f1B[i] + fj, 0.f);   // f1B[i] uniform per wave -> s_load
    acc += (a > 0) ? __expf(e) : 0.f;
  }

  __shared__ float red[4][64];
  red[rq][col] = acc;
  __syncthreads();
  if (tid < 64)
    s_out[b * NNODE + j0 + tid] =
        red[0][tid] + red[1][tid] + red[2][tid] + red[3][tid];
}

// ---------------------------------------------------------------------------
// Kernel 4: fused softmax-apply + GEMM + ELU.
// h'[i,o] = ELU( sum_j P[i,j] * Wh[j,o] ), P regenerated on the fly per 64x32
// tile (adj mask + exp + 1/s_j), staged bf16 in LDS, consumed by WMMA.
// The 134 MB attention matrix never touches HBM.
// Block: 64 rows x 256 feats; K (=j) stepped by 32.
// ---------------------------------------------------------------------------
__global__ void __launch_bounds__(256)
attn_gemm_kernel(const int* __restrict__ adj, const float* __restrict__ f1,
                 const float* __restrict__ f2, const float* __restrict__ s,
                 const __bf16* __restrict__ wh_t, float* __restrict__ out)
{
  const int b    = blockIdx.y;
  const int i0   = blockIdx.x * 64;
  const int tid  = threadIdx.x;
  const int lane = tid & 31;
  const int wave = tid >> 5;
  const int rg   = wave & 3;
  const int fg   = wave >> 2;

  __shared__ __align__(16) __bf16 ptile[64][40];  // P tile (padded: 80B rows)
  __shared__ float f1s[64];

  if (tid < 64) f1s[tid] = f1[b * NNODE + i0 + tid];

  v8f acc[8] = {};

  // P-producer mapping: 8 rows x 32 cols per pass, 8 passes
  const int pcol  = tid & 31;
  const int prow0 = tid >> 5;
  // A-fragment mapping (from LDS P tile)
  const int arow = rg * 16 + (lane & 15);
  const int akb  = (lane < 16) ? 0 : 8;
  // B-fragment mapping (from wh_t)
  const int bo   = fg * 128 + (lane & 15);
  const int bkb  = (lane < 16) ? 0 : 8;

  const int*    adjB = adj  + (size_t)b * NNODE * NNODE;
  const __bf16* whB  = wh_t + (size_t)b * FDIM * NNODE;
  const float*  f2B  = f2 + b * NNODE;
  const float*  sB   = s  + b * NNODE;

  for (int j0 = 0; j0 < NNODE; j0 += 32) {
    // per-column softmax scalars
    const float fj  = f2B[j0 + pcol];
    const float sj  = sB[j0 + pcol];
    const float inv = (sj > 0.f) ? (1.0f / sj) : 0.f;
    const float fb  = (sj > 0.f) ? 0.f : (1.0f / (float)NNODE);  // all-masked col

    if (j0 + 32 < NNODE)  // global_prefetch_b8 of next adj tile (one line/row)
      __builtin_prefetch(adjB + (size_t)(i0 + (tid >> 2)) * NNODE + j0 + 32, 0, 1);

    __syncthreads();  // previous iteration's LDS readers are done
#pragma unroll
    for (int m = 0; m < 8; ++m) {
      const int r = prow0 + m * 8;
      const int a = adjB[(size_t)(i0 + r) * NNODE + j0 + pcol];
      const float e = fmaxf(f1s[r] + fj, 0.f);
      const float p = ((a > 0) ? __expf(e) * inv : 0.f) + fb;
      ptile[r][pcol] = (__bf16)p;
    }
    __syncthreads();

    v16bf afrag = ld_bf16_8x2(&ptile[arow][akb], &ptile[arow][16 + akb]);
#pragma unroll
    for (int ct = 0; ct < 8; ++ct) {
      const __bf16* pb = whB + (size_t)(bo + ct * 16) * NNODE + j0 + bkb;
      v16bf bfrag = ld_bf16_8x2(pb, pb + 16);  // two 8-element K runs (interleaved)
      acc[ct] = wmma_bf16(afrag, bfrag, acc[ct]);
    }
  }

  // ELU epilogue + store (f32 out, [b][i][o])
  float* outB = out + (size_t)b * NNODE * FDIM;
#pragma unroll
  for (int ct = 0; ct < 8; ++ct) {
    const int col   = fg * 128 + ct * 16 + (lane & 15);
    const int rbase = i0 + rg * 16 + 8 * (lane >> 4);
#pragma unroll
    for (int v = 0; v < 8; ++v) {
      const float x = acc[ct][v];
      outB[(size_t)(rbase + v) * FDIM + col] = (x > 0.f) ? x : (__expf(x) - 1.f);
    }
  }
}

// ---------------------------------------------------------------------------
extern "C" void kernel_launch(void* const* d_in, const int* in_sizes, int n_in,
                              void* d_out, int out_size, void* d_ws, size_t ws_size,
                              hipStream_t stream) {
  const float* h   = (const float*)d_in[0];   // [8,2048,256] f32
  const int*   adj = (const int*)d_in[1];     // [8,2048,2048] i32
  const float* Ww  = (const float*)d_in[2];   // [256,256]
  const float* Wb  = (const float*)d_in[3];   // [256]
  const float* a1w = (const float*)d_in[4];   // [1,256]
  const float* a1b = (const float*)d_in[5];   // [1]
  const float* a2w = (const float*)d_in[6];   // [1,256]
  const float* a2b = (const float*)d_in[7];   // [1]
  float* out = (float*)d_out;                 // [8,2048,256] f32

  // workspace layout: wh_t (bf16, 8.4MB) | f1 | f2 | s | Wbf (bf16, 128KB)
  // (all fully written before being read; no dependence on initial contents)
  __bf16* wh_t = (__bf16*)d_ws;
  float* f1 = (float*)((char*)d_ws + (size_t)BATCH * FDIM * NNODE * sizeof(__bf16));
  float* f2 = f1 + BATCH * NNODE;
  float* s  = f2 + BATCH * NNODE;
  __bf16* Wbf = (__bf16*)(s + BATCH * NNODE);

  wcvt_kernel     <<<dim3(FDIM * FDIM / 1024),     256, 0, stream>>>(Ww, Wbf);
  wh_gemm_kernel  <<<dim3(NNODE / 64,  BATCH), 256, 0, stream>>>(h, Wbf, Wb, wh_t);
  f12_kernel      <<<dim3(NNODE / 256, BATCH), 256, 0, stream>>>(wh_t, a1w, a1b,
                                                                 a2w, a2b, f1, f2);
  stats_kernel    <<<dim3(NNODE / 64,  BATCH), 256, 0, stream>>>(adj, f1, f2, s);
  attn_gemm_kernel<<<dim3(NNODE / 64,  BATCH), 256, 0, stream>>>(adj, f1, f2, s,
                                                                 wh_t, out);
}
